// MambaBlock_81192061763613
// MI455X (gfx1250) — matrix operand order repaired and checked by reading, not compiled, not run on hardware
//
#include <hip/hip_runtime.h>
#include <hip/hip_bf16.h>

// Problem dims
#define B_   8
#define S_   2048
#define IN_  1024
#define H_   2048
#define N_   32
#define KC_  4
#define M_   (B_ * S_)      // 16384 rows
#define CHUNK 64
#define NCHUNK (S_ / CHUNK) // 32

// LDS tile geometry: block tile 64(M) x 128(N), K-stage = 64
#define KSTAGE    64
#define LDSTRIDE  72        // bf16 elems per LDS row (144 B, staggers banks, 16B-aligned)

typedef __bf16 bf16x16 __attribute__((ext_vector_type(16)));
typedef __bf16 bf16x8  __attribute__((ext_vector_type(8)));
typedef float  f32x8   __attribute__((ext_vector_type(8)));

__device__ __forceinline__ float silu(float x) { return x / (1.f + __expf(-x)); }

// ---------------------------------------------------------------------------
// Async global -> LDS staging (gfx1250 ASYNCcnt path).
// A tile: 64 rows x 64 k ; B tile: 128 rows x 64 k ; both row-major in LDS
// with LDSTRIDE padding. 256 threads issue 2 (A) + 4 (B) b128 async loads.
// ---------------------------------------------------------------------------
__device__ __forceinline__ void async_fill_tile(
    const __bf16* __restrict__ Ag, const __bf16* __restrict__ Bg,
    int lda, int ldb, int mblk, int nblk, int kbase,
    uint32_t aLds, uint32_t bLds, int tid) {
#pragma unroll
  for (int it = 0; it < 2; ++it) {          // A: 512 chunks of 16B
    int c = tid + it * 256;
    int row = c >> 3, kq = (c & 7) << 3;    // 8 chunks per 64-elem row
    uint64_t g = (uint64_t)(uintptr_t)(Ag + (size_t)(mblk + row) * lda + kbase + kq);
    uint32_t l = aLds + row * (LDSTRIDE * 2) + (kq << 1);
    asm volatile("global_load_async_to_lds_b128 %0, %1, off"
                 :: "v"(l), "v"(g) : "memory");
  }
#pragma unroll
  for (int it = 0; it < 4; ++it) {          // B: 1024 chunks of 16B
    int c = tid + it * 256;
    int row = c >> 3, kq = (c & 7) << 3;
    uint64_t g = (uint64_t)(uintptr_t)(Bg + (size_t)(nblk + row) * ldb + kbase + kq);
    uint32_t l = bLds + row * (LDSTRIDE * 2) + (kq << 1);
    asm volatile("global_load_async_to_lds_b128 %0, %1, off"
                 :: "v"(l), "v"(g) : "memory");
  }
}

__device__ __forceinline__ void wait_async_and_barrier() {
  asm volatile("s_wait_asynccnt 0x0" ::: "memory");
  __syncthreads();
}

// ---------------------------------------------------------------------------
// WMMA fragment loads from LDS (layouts per CDNA5 ISA 7.12.2)
// A 16x32: lanes 0-15 -> K[0..7],K[16..23]; lanes 16-31 -> K[8..15],K[24..31]
// B 32x16: col = lane&15 (weight row), same K split as A
// ---------------------------------------------------------------------------
__device__ __forceinline__ bf16x16 lds_a_frag(const __bf16* sA, int m0l, int ks, int lane) {
  const __bf16* p = sA + (m0l + (lane & 15)) * LDSTRIDE + ks + ((lane >> 4) << 3);
  bf16x8 lo = *(const bf16x8*)(p);
  bf16x8 hi = *(const bf16x8*)(p + 16);
  bf16x16 v;
#pragma unroll
  for (int i = 0; i < 8; ++i) { v[i] = lo[i]; v[8 + i] = hi[i]; }
  return v;
}

__device__ __forceinline__ bf16x16 lds_b_frag(const __bf16* sB, int n0l, int ks, int lane) {
  const __bf16* p = sB + (n0l + (lane & 15)) * LDSTRIDE + ks + ((lane >> 4) << 3);
  bf16x8 lo = *(const bf16x8*)(p);
  bf16x8 hi = *(const bf16x8*)(p + 16);
  bf16x16 v;
#pragma unroll
  for (int i = 0; i < 8; ++i) { v[i] = lo[i]; v[8 + i] = hi[i]; }
  return v;
}

// One K-stage of compute: 2 sub-steps of K=32, 4 WMMAs each (32x32 per wave)
__device__ __forceinline__ void compute_stage(const __bf16* sA, const __bf16* sB,
                                              int wm, int wn, int lane, f32x8 acc[2][2]) {
#pragma unroll
  for (int ks = 0; ks < KSTAGE; ks += 32) {
    bf16x16 a0 = lds_a_frag(sA, wm * 32,      ks, lane);
    bf16x16 a1 = lds_a_frag(sA, wm * 32 + 16, ks, lane);
    bf16x16 b0 = lds_b_frag(sB, wn * 32,      ks, lane);
    bf16x16 b1 = lds_b_frag(sB, wn * 32 + 16, ks, lane);
    acc[0][0] = __builtin_amdgcn_wmma_f32_16x16x32_bf16(false, a0, false, b0, (short)0, acc[0][0], false, false);
    acc[0][1] = __builtin_amdgcn_wmma_f32_16x16x32_bf16(false, a0, false, b1, (short)0, acc[0][1], false, false);
    acc[1][0] = __builtin_amdgcn_wmma_f32_16x16x32_bf16(false, a1, false, b0, (short)0, acc[1][0], false, false);
    acc[1][1] = __builtin_amdgcn_wmma_f32_16x16x32_bf16(false, a1, false, b1, (short)0, acc[1][1], false, false);
  }
}

// ---------------------------------------------------------------------------
// fp32 -> bf16 conversion
// ---------------------------------------------------------------------------
__global__ void cvt_f32_bf16_kernel(const float* __restrict__ in,
                                    __hip_bfloat16* __restrict__ out, size_t n) {
  size_t i = (size_t)blockIdx.x * blockDim.x + threadIdx.x;
  if (i < n) out[i] = __float2bfloat16(in[i]);
}

// ---------------------------------------------------------------------------
// GEMM1: xz = x @ W_in^T + b_in ; split into xh_raw (f32) and silu(gate) bf16
// Double-buffered async LDS staging; wave -> 32x32 output tile.
// ---------------------------------------------------------------------------
__global__ __launch_bounds__(256) void gemm_xz_kernel(
    const __hip_bfloat16* __restrict__ xb,    // [M, IN] bf16
    const __hip_bfloat16* __restrict__ wib,   // [2H, IN] bf16
    const float* __restrict__ b_in,           // [2H]
    float* __restrict__ xh_raw,               // [M, H]
    __hip_bfloat16* __restrict__ gate_act) {  // [M, H] = silu(gate)
  __shared__ __bf16 sA[2][64 * LDSTRIDE];
  __shared__ __bf16 sB[2][128 * LDSTRIDE];
  const int tid = threadIdx.x;
  const int lane = tid & 31;
  const int wid  = tid >> 5;
  const int wm = wid >> 2, wn = wid & 3;
  const int mblk = blockIdx.y * 64;
  const int nblk = blockIdx.x * 128;
  const __bf16* Ag = reinterpret_cast<const __bf16*>(xb);
  const __bf16* Bg = reinterpret_cast<const __bf16*>(wib);
  uint32_t aL[2] = {(uint32_t)(uintptr_t)&sA[0][0], (uint32_t)(uintptr_t)&sA[1][0]};
  uint32_t bL[2] = {(uint32_t)(uintptr_t)&sB[0][0], (uint32_t)(uintptr_t)&sB[1][0]};

  f32x8 acc[2][2] = {{{}, {}}, {{}, {}}};
  const int NST = IN_ / KSTAGE;   // 16

  async_fill_tile(Ag, Bg, IN_, IN_, mblk, nblk, 0, aL[0], bL[0], tid);
  wait_async_and_barrier();
  for (int st = 0; st < NST; ++st) {
    if (st + 1 < NST)
      async_fill_tile(Ag, Bg, IN_, IN_, mblk, nblk, (st + 1) * KSTAGE,
                      aL[(st + 1) & 1], bL[(st + 1) & 1], tid);
    compute_stage(&sA[st & 1][0], &sB[st & 1][0], wm, wn, lane, acc);
    wait_async_and_barrier();
  }

  const int m0 = mblk + wm * 32;
  const int n0 = nblk + wn * 32;
  const int rowbase = m0 + ((lane >> 4) << 3);
  const int colbase = lane & 15;
#pragma unroll
  for (int i = 0; i < 2; ++i)
#pragma unroll
    for (int j = 0; j < 2; ++j) {
      int n = n0 + j * 16 + colbase;
      float bias = b_in[n];
#pragma unroll
      for (int r = 0; r < 8; ++r) {
        int row = rowbase + i * 16 + r;
        float v = acc[i][j][r] + bias;
        if (n < H_) xh_raw[(size_t)row * H_ + n] = v;
        else        gate_act[(size_t)row * H_ + (n - H_)] = __float2bfloat16(silu(v));
      }
    }
}

// ---------------------------------------------------------------------------
// GEMM3: out = y_in @ W_out^T + b_out + x  (residual fused, fp32 out)
// ---------------------------------------------------------------------------
__global__ __launch_bounds__(256) void gemm_out_kernel(
    const __hip_bfloat16* __restrict__ y_in,  // [M, H] bf16
    const __hip_bfloat16* __restrict__ wob,   // [IN, H] bf16
    const float* __restrict__ b_out,          // [IN]
    const float* __restrict__ x_resid,        // [M, IN]
    float* __restrict__ out) {                // [M, IN]
  __shared__ __bf16 sA[2][64 * LDSTRIDE];
  __shared__ __bf16 sB[2][128 * LDSTRIDE];
  const int tid = threadIdx.x;
  const int lane = tid & 31;
  const int wid  = tid >> 5;
  const int wm = wid >> 2, wn = wid & 3;
  const int mblk = blockIdx.y * 64;
  const int nblk = blockIdx.x * 128;
  const __bf16* Ag = reinterpret_cast<const __bf16*>(y_in);
  const __bf16* Bg = reinterpret_cast<const __bf16*>(wob);
  uint32_t aL[2] = {(uint32_t)(uintptr_t)&sA[0][0], (uint32_t)(uintptr_t)&sA[1][0]};
  uint32_t bL[2] = {(uint32_t)(uintptr_t)&sB[0][0], (uint32_t)(uintptr_t)&sB[1][0]};

  f32x8 acc[2][2] = {{{}, {}}, {{}, {}}};
  const int NST = H_ / KSTAGE;    // 32

  async_fill_tile(Ag, Bg, H_, H_, mblk, nblk, 0, aL[0], bL[0], tid);
  wait_async_and_barrier();
  for (int st = 0; st < NST; ++st) {
    if (st + 1 < NST)
      async_fill_tile(Ag, Bg, H_, H_, mblk, nblk, (st + 1) * KSTAGE,
                      aL[(st + 1) & 1], bL[(st + 1) & 1], tid);
    compute_stage(&sA[st & 1][0], &sB[st & 1][0], wm, wn, lane, acc);
    wait_async_and_barrier();
  }

  const int m0 = mblk + wm * 32;
  const int n0 = nblk + wn * 32;
  const int rowbase = m0 + ((lane >> 4) << 3);
  const int colbase = lane & 15;
#pragma unroll
  for (int i = 0; i < 2; ++i)
#pragma unroll
    for (int j = 0; j < 2; ++j) {
      int n = n0 + j * 16 + colbase;
      float bias = b_out[n];
#pragma unroll
      for (int r = 0; r < 8; ++r) {
        int row = rowbase + i * 16 + r;
        size_t o = (size_t)row * IN_ + n;
        out[o] = acc[i][j][r] + bias + x_resid[o];
      }
    }
}

// ---------------------------------------------------------------------------
// Causal depthwise conv1d (K=4) + bias + SiLU
// ---------------------------------------------------------------------------
__global__ void conv_silu_kernel(const float* __restrict__ xh_raw,
                                 const float* __restrict__ conv_w,
                                 const float* __restrict__ conv_b,
                                 float* __restrict__ xh_act) {
  size_t idx = (size_t)blockIdx.x * blockDim.x + threadIdx.x;
  if (idx >= (size_t)M_ * H_) return;
  int h = (int)(idx & (H_ - 1));
  int m = (int)(idx >> 11);      // H = 2048 = 2^11
  int s = m & (S_ - 1);
  float acc = conv_b[h];
#pragma unroll
  for (int j = 0; j < KC_; ++j)
    if (s - j >= 0) acc += conv_w[h * KC_ + (KC_ - 1 - j)] * xh_raw[idx - (size_t)j * H_];
  xh_act[idx] = silu(acc);
}

// ---------------------------------------------------------------------------
// x_ssm = xh_act @ W_ssm^T + b_ssm  (N=32 -> memory bound, plain fp32)
// ---------------------------------------------------------------------------
__global__ __launch_bounds__(256) void gemm_ssm_kernel(
    const float* __restrict__ xh_act, const float* __restrict__ W_ssm,
    const float* __restrict__ b_ssm, float* __restrict__ x_ssm) {
  __shared__ float red[8][N_];
  int m = blockIdx.x;
  int n = threadIdx.x & 31;
  int part = threadIdx.x >> 5;
  const float* xr = xh_act + (size_t)m * H_ + part * (H_ / 8);
  const float* wr = W_ssm + (size_t)n * H_ + part * (H_ / 8);
  float sum = 0.f;
#pragma unroll 8
  for (int i = 0; i < H_ / 8; ++i) sum += xr[i] * wr[i];
  red[part][n] = sum;
  __syncthreads();
  if (part == 0) {
    float t = 0.f;
#pragma unroll
    for (int p = 0; p < 8; ++p) t += red[p][n];
    x_ssm[(size_t)m * N_ + n] = t + b_ssm[n];
  }
}

// ---------------------------------------------------------------------------
// A^64 via 6 LDS squarings (32x32, one wave)
// ---------------------------------------------------------------------------
__global__ void matpow64_kernel(const float* __restrict__ A, float* __restrict__ A64) {
  __shared__ float Ma[N_ * N_], Mb[N_ * N_];
  int j = threadIdx.x;
  for (int i = 0; i < N_; ++i) Ma[i * N_ + j] = A[i * N_ + j];
  __syncthreads();
  for (int it = 0; it < 6; ++it) {
    for (int i = 0; i < N_; ++i) {
      float t = 0.f;
      for (int k = 0; k < N_; ++k) t += Ma[i * N_ + k] * Ma[k * N_ + j];
      Mb[i * N_ + j] = t;
    }
    __syncthreads();
    for (int i = 0; i < N_; ++i) Ma[i * N_ + j] = Mb[i * N_ + j];
    __syncthreads();
  }
  for (int i = 0; i < N_; ++i) A64[i * N_ + j] = Ma[i * N_ + j];
}

// ---------------------------------------------------------------------------
// Chunked affine scan, phase 1: local end-state from zero init per chunk
// ---------------------------------------------------------------------------
__global__ void scan_pass1_kernel(const float* __restrict__ x_ssm,
                                  const float* __restrict__ A,
                                  const float* __restrict__ Bv,
                                  float* __restrict__ V) {
  int b = blockIdx.x >> 5, c = blockIdx.x & 31, lane = threadIdx.x;
  float a[N_];
#pragma unroll
  for (int j = 0; j < N_; ++j) a[j] = A[lane * N_ + j];
  float bv = Bv[lane];
  __shared__ float hs[N_];
  float h = 0.f;
  const float* u = x_ssm + ((size_t)(b * S_ + c * CHUNK)) * N_ + lane;
  for (int t = 0; t < CHUNK; ++t) {
    hs[lane] = h;
    __syncthreads();
    float nh = 0.f;
#pragma unroll
    for (int j = 0; j < N_; ++j) nh += a[j] * hs[j];
    __syncthreads();
    h = nh + u[t * N_] * bv;
  }
  V[(size_t)blockIdx.x * N_ + lane] = h;
}

// Phase 2: serial combine over chunks (depth 32) -> chunk start states
__global__ void scan_combine_kernel(const float* __restrict__ A64,
                                    const float* __restrict__ V,
                                    float* __restrict__ Hs) {
  int b = blockIdx.x, lane = threadIdx.x;
  float a[N_];
#pragma unroll
  for (int j = 0; j < N_; ++j) a[j] = A64[lane * N_ + j];
  __shared__ float hs[N_];
  float h = 0.f;
  for (int c = 0; c < NCHUNK; ++c) {
    size_t off = ((size_t)b * NCHUNK + c) * N_;
    Hs[off + lane] = h;
    hs[lane] = h;
    __syncthreads();
    float nh = 0.f;
#pragma unroll
    for (int j = 0; j < N_; ++j) nh += a[j] * hs[j];
    __syncthreads();
    h = nh + V[off + lane];
  }
}

// Phase 3: replay chunk with true start state, emit s_t = Cv . h_t
__global__ void scan_pass2_kernel(const float* __restrict__ x_ssm,
                                  const float* __restrict__ A,
                                  const float* __restrict__ Bv,
                                  const float* __restrict__ Cv,
                                  const float* __restrict__ Hs,
                                  float* __restrict__ s_out) {
  int b = blockIdx.x >> 5, c = blockIdx.x & 31, lane = threadIdx.x;
  float a[N_];
#pragma unroll
  for (int j = 0; j < N_; ++j) a[j] = A[lane * N_ + j];
  float bv = Bv[lane], cv = Cv[lane];
  __shared__ float hs[N_];
  float h = Hs[(size_t)blockIdx.x * N_ + lane];
  const float* u = x_ssm + ((size_t)(b * S_ + c * CHUNK)) * N_ + lane;
  for (int t = 0; t < CHUNK; ++t) {
    hs[lane] = h;
    __syncthreads();
    float nh = 0.f;
#pragma unroll
    for (int j = 0; j < N_; ++j) nh += a[j] * hs[j];
    __syncthreads();
    h = nh + u[t * N_] * bv;
    float sv = h * cv;
#pragma unroll
    for (int off = 16; off; off >>= 1) sv += __shfl_xor(sv, off, 32);
    if (lane == 0) s_out[(size_t)b * S_ + c * CHUNK + t] = sv;
  }
}

// ---------------------------------------------------------------------------
// y_in = (s + Dv) * xh_act * silu(gate)  -> bf16 for final GEMM
// ---------------------------------------------------------------------------
__global__ void yprep_kernel(const float* __restrict__ s, const float* __restrict__ Dv,
                             const float* __restrict__ xh_act,
                             const __hip_bfloat16* __restrict__ gate_act,
                             __hip_bfloat16* __restrict__ y_in) {
  size_t idx = (size_t)blockIdx.x * blockDim.x + threadIdx.x;
  if (idx >= (size_t)M_ * H_) return;
  int h = (int)(idx & (H_ - 1));
  size_t m = idx >> 11;
  float v = (s[m] + Dv[h]) * xh_act[idx] * __bfloat162float(gate_act[idx]);
  y_in[idx] = __float2bfloat16(v);
}

// ---------------------------------------------------------------------------
extern "C" void kernel_launch(void* const* d_in, const int* in_sizes, int n_in,
                              void* d_out, int out_size, void* d_ws, size_t ws_size,
                              hipStream_t stream) {
  const float* x      = (const float*)d_in[0];
  const float* W_in   = (const float*)d_in[1];
  const float* b_in   = (const float*)d_in[2];
  const float* conv_w = (const float*)d_in[3];
  const float* conv_b = (const float*)d_in[4];
  const float* W_ssm  = (const float*)d_in[5];
  const float* b_ssm  = (const float*)d_in[6];
  const float* A      = (const float*)d_in[7];
  const float* Bv     = (const float*)d_in[8];
  const float* Cv     = (const float*)d_in[9];
  const float* Dv     = (const float*)d_in[10];
  const float* W_out  = (const float*)d_in[11];
  const float* b_out  = (const float*)d_in[12];
  float* out = (float*)d_out;

  char* wsp = (char*)d_ws;
  auto alloc = [&](size_t bytes) {
    char* p = wsp;
    wsp += (bytes + 255) & ~(size_t)255;
    return p;
  };
  __hip_bfloat16* xb       = (__hip_bfloat16*)alloc((size_t)M_ * IN_ * 2);
  __hip_bfloat16* wib      = (__hip_bfloat16*)alloc((size_t)2 * H_ * IN_ * 2);
  __hip_bfloat16* wob      = (__hip_bfloat16*)alloc((size_t)IN_ * H_ * 2);
  float*          xh_raw   = (float*)alloc((size_t)M_ * H_ * 4);
  __hip_bfloat16* gate_act = (__hip_bfloat16*)alloc((size_t)M_ * H_ * 2);
  float*          xh_act   = (float*)alloc((size_t)M_ * H_ * 4);
  float*          x_ssm    = (float*)alloc((size_t)M_ * N_ * 4);
  float*          s_buf    = (float*)alloc((size_t)M_ * 4);
  __hip_bfloat16* y_in     = (__hip_bfloat16*)alloc((size_t)M_ * H_ * 2);
  float*          A64      = (float*)alloc((size_t)N_ * N_ * 4);
  float*          Vbuf     = (float*)alloc((size_t)B_ * NCHUNK * N_ * 4);
  float*          Hsbuf    = (float*)alloc((size_t)B_ * NCHUNK * N_ * 4);

  // 1. fp32 -> bf16 conversions
  size_t nx = (size_t)M_ * IN_;
  cvt_f32_bf16_kernel<<<(unsigned)((nx + 255) / 256), 256, 0, stream>>>(x, xb, nx);
  size_t nwi = (size_t)2 * H_ * IN_;
  cvt_f32_bf16_kernel<<<(unsigned)((nwi + 255) / 256), 256, 0, stream>>>(W_in, wib, nwi);
  size_t nwo = (size_t)IN_ * H_;
  cvt_f32_bf16_kernel<<<(unsigned)((nwo + 255) / 256), 256, 0, stream>>>(W_out, wob, nwo);

  // 2. xz GEMM (WMMA, async-LDS double buffered) with split/SiLU epilogue
  gemm_xz_kernel<<<dim3(2 * H_ / 128, M_ / 64), 256, 0, stream>>>(xb, wib, b_in, xh_raw, gate_act);

  // 3. causal depthwise conv + SiLU
  size_t nmh = (size_t)M_ * H_;
  conv_silu_kernel<<<(unsigned)((nmh + 255) / 256), 256, 0, stream>>>(xh_raw, conv_w, conv_b, xh_act);

  // 4. small SSM projection GEMM (fp32)
  gemm_ssm_kernel<<<M_, 256, 0, stream>>>(xh_act, W_ssm, b_ssm, x_ssm);

  // 5. chunked parallel affine scan
  matpow64_kernel<<<1, 32, 0, stream>>>(A, A64);
  scan_pass1_kernel<<<B_ * NCHUNK, 32, 0, stream>>>(x_ssm, A, Bv, Vbuf);
  scan_combine_kernel<<<B_, 32, 0, stream>>>(A64, Vbuf, Hsbuf);
  scan_pass2_kernel<<<B_ * NCHUNK, 32, 0, stream>>>(x_ssm, A, Bv, Cv, Hsbuf, s_buf);

  // 6. fused elementwise (ssm_out * silu(gate)) -> bf16
  yprep_kernel<<<(unsigned)((nmh + 255) / 256), 256, 0, stream>>>(s_buf, Dv, xh_act, gate_act, y_in);

  // 7. output GEMM (WMMA, async-LDS double buffered) with bias + residual
  gemm_out_kernel<<<dim3(IN_ / 128, M_ / 64), 256, 0, stream>>>(y_in, wob, b_out, x, out);
}
